// ConformerEncoderLayer_10033043603610
// MI455X (gfx1250) — matrix operand order repaired and checked by reading, not compiled, not run on hardware
//
#include <hip/hip_runtime.h>
#include <hip/hip_bf16.h>
#include <math.h>

typedef unsigned short u16;
typedef __attribute__((ext_vector_type(16))) __bf16 bf16x16;
typedef __attribute__((ext_vector_type(8)))  float  v8f;
typedef __attribute__((ext_vector_type(4)))  unsigned int u32x4;

namespace conf {
constexpr int T   = 512;
constexpr int B   = 8;
constexpr int D   = 512;
constexpr int H   = 8;
constexpr int DFF = 2048;
constexpr int HD  = 64;
constexpr int KC  = 31;
constexpr int NR  = T * B;       // 4096 token rows
constexpr int BH  = B * H;       // 64
constexpr int TP  = 2 * T - 1;   // 1023

constexpr size_t al(size_t x) { return (x + 255) & ~size_t(255); }
// ---- workspace byte offsets (total ~216 MB) ----
constexpr size_t WB_FFM1 = 0;
constexpr size_t WB_FFM2 = al(WB_FFM1 + size_t(DFF) * D * 2);
constexpr size_t WB_FF1  = al(WB_FFM2 + size_t(D) * DFF * 2);
constexpr size_t WB_FF2  = al(WB_FF1  + size_t(DFF) * D * 2);
constexpr size_t WB_INP  = al(WB_FF2  + size_t(D) * DFF * 2);
constexpr size_t WB_OUT  = al(WB_INP  + size_t(3 * D) * D * 2);
constexpr size_t WB_POS  = al(WB_OUT  + size_t(D) * D * 2);
constexpr size_t WB_PW1  = al(WB_POS  + size_t(D) * D * 2);
constexpr size_t WB_PW2  = al(WB_PW1  + size_t(2 * D) * D * 2);
constexpr size_t PE_BF   = al(WB_PW2  + size_t(D) * D * 2);     // 1024 rows (row 1023 zeroed)
constexpr size_t RES     = al(PE_BF   + size_t(1024) * D * 2);
constexpr size_t XN      = al(RES     + size_t(NR) * D * 4);
constexpr size_t HID     = al(XN      + size_t(NR) * D * 2);
constexpr size_t QKV     = al(HID     + size_t(NR) * DFF * 2);
constexpr size_t QU      = al(QKV     + size_t(NR) * 3 * D * 4);
constexpr size_t QV      = al(QU      + size_t(BH) * T * HD * 2);
constexpr size_t KB      = al(QV      + size_t(BH) * T * HD * 2);
constexpr size_t VT      = al(KB      + size_t(BH) * T * HD * 2);
constexpr size_t PBF     = al(VT      + size_t(BH) * T * HD * 2); // 1024 rows (row 1023 = 0)
constexpr size_t SC      = al(PBF     + size_t(1024) * D * 2);
constexpr size_t AT      = al(SC      + size_t(BH) * T * T * 4);
constexpr size_t CTX     = al(AT      + size_t(BH) * T * T * 2);
constexpr size_t PW1O    = al(CTX     + size_t(NR) * D * 2);
constexpr size_t GLUO    = al(PW1O    + size_t(NR) * 2 * D * 4);
} // namespace conf

// ---------------- device helpers ----------------

__device__ __forceinline__ u16 f2bf(float f) {
  unsigned int x = __float_as_uint(f);
  x += 0x7FFFu + ((x >> 16) & 1u);            // round-to-nearest-even
  return (u16)(x >> 16);
}
__device__ __forceinline__ float sigm(float x) { return 1.f / (1.f + __expf(-x)); }

// Fragment loads; p already points at (row, k0) for this lane.
// A fragment (16x32): 16 contiguous k per lane (two adjacent b128 loads)
__device__ __forceinline__ bf16x16 ldA(const u16* p) {
  union { bf16x16 v; u32x4 q[2]; } f;
  f.q[0] = *(const u32x4*)(p);
  f.q[1] = *(const u32x4*)(p + 8);
  return f.v;
}
// B fragment (32x16): 16 contiguous k per lane (two adjacent b128 loads)
__device__ __forceinline__ bf16x16 ldB(const u16* p) {
  union { bf16x16 v; u32x4 q[2]; } f;
  f.q[0] = *(const u32x4*)(p);
  f.q[1] = *(const u32x4*)(p + 8);
  return f.v;
}

#define WMMA_BF16(A_, B_, C_) \
  __builtin_amdgcn_wmma_f32_16x16x32_bf16(false, (A_), false, (B_), (short)0, (C_), false, false)

// Wave-level 32x64 tile of C = A(MxK) @ B(NxK)^T.
// 8 f32 accumulators, 2 A frags, 4 B frags -> 8 WMMAs per 6 fragment loads.
// All operand shapes are tile-aligned (padded where needed) -> branch-free inner loop.
__device__ __forceinline__ void wave_gemm_32x64(const u16* A, int lda, int m0,
                                                const u16* Bm, int ldb, int n0,
                                                int K, v8f acc[8]) {
  const int lane = threadIdx.x & 31;
  const int lr  = lane & 15;
  const int kbA = (lane >> 4) << 4;
  const int kbB = (lane >> 4) << 4;
  const u16* pa0 = A + (long)(m0 + lr) * lda + kbA;
  const u16* pa1 = A + (long)(m0 + 16 + lr) * lda + kbA;
  const u16* pb0 = Bm + (long)(n0 + lr) * ldb + kbB;
  const u16* pb1 = Bm + (long)(n0 + 16 + lr) * ldb + kbB;
  const u16* pb2 = Bm + (long)(n0 + 32 + lr) * ldb + kbB;
  const u16* pb3 = Bm + (long)(n0 + 48 + lr) * ldb + kbB;
  for (int kk = 0; kk < K; kk += 32) {
    bf16x16 a0 = ldA(pa0);
    bf16x16 a1 = ldA(pa1);
    bf16x16 b0 = ldB(pb0);
    bf16x16 b1 = ldB(pb1);
    bf16x16 b2 = ldB(pb2);
    bf16x16 b3 = ldB(pb3);
    acc[0] = WMMA_BF16(a0, b0, acc[0]);
    acc[1] = WMMA_BF16(a0, b1, acc[1]);
    acc[2] = WMMA_BF16(a0, b2, acc[2]);
    acc[3] = WMMA_BF16(a0, b3, acc[3]);
    acc[4] = WMMA_BF16(a1, b0, acc[4]);
    acc[5] = WMMA_BF16(a1, b1, acc[5]);
    acc[6] = WMMA_BF16(a1, b2, acc[6]);
    acc[7] = WMMA_BF16(a1, b3, acc[7]);
    pa0 += 32; pa1 += 32; pb0 += 32; pb1 += 32; pb2 += 32; pb3 += 32;
  }
}

// ---------------- elementwise kernels ----------------

__global__ void f32_to_bf16_kernel(const float* in, u16* out, int n) {
  int i = blockIdx.x * 256 + threadIdx.x;
  if (i < n) out[i] = f2bf(in[i]);
}

__global__ void zero_u16_kernel(u16* out, int n) {
  int i = blockIdx.x * 256 + threadIdx.x;
  if (i < n) out[i] = 0;
}

__global__ void copy_f32_kernel(const float* in, float* out, int n) {
  int i = blockIdx.x * 256 + threadIdx.x;
  if (i < n) out[i] = in[i];
}

// wave-per-row layernorm over 512 columns
template <int OUTBF>
__global__ void layernorm_kernel(const float* x, const float* g, const float* bb,
                                 u16* outB, float* outF, int rows) {
  int wid = threadIdx.x >> 5, lane = threadIdx.x & 31;
  int row = blockIdx.x * 8 + wid;
  if (row >= rows) return;
  const float* xr = x + (long)row * 512;
  float v[16], s = 0.f, sq = 0.f;
#pragma unroll
  for (int i = 0; i < 16; i++) { v[i] = xr[lane + 32 * i]; s += v[i]; sq += v[i] * v[i]; }
#pragma unroll
  for (int m = 16; m >= 1; m >>= 1) { s += __shfl_xor(s, m, 32); sq += __shfl_xor(sq, m, 32); }
  float mu = s * (1.f / 512.f);
  float var = sq * (1.f / 512.f) - mu * mu;
  float rs = rsqrtf(var + 1e-5f);
#pragma unroll
  for (int i = 0; i < 16; i++) {
    int c = lane + 32 * i;
    float o = (v[i] - mu) * rs * g[c] + bb[c];
    if (OUTBF) outB[(long)row * 512 + c] = f2bf(o);
    else       outF[(long)row * 512 + c] = o;
  }
}

// ---------------- generic GEMM: C = act(A @ W^T + bias) ----------------
// EPI: 0 = write f32, 1 = write bf16, 2 = res += alpha*val.  ACT: 0 none, 1 swish.
// BIAS: 0/1 compile-time. M multiple of 128 (padded), N multiple of 128.
// Block: 8 waves = 128(M) x 128(N) tile; wave = 32x64.
template <int EPI, int ACT, int BIAS>
__global__ void __launch_bounds__(256)
gemm_kernel(const u16* A, int lda, const u16* W, int ldb, const float* bias,
            void* outp, float* res, int N, int K, float alpha) {
  const int lane = threadIdx.x & 31, w = threadIdx.x >> 5;
  const int m0 = blockIdx.y * 128 + (w & 3) * 32;
  const int n0 = blockIdx.x * 128 + (w >> 2) * 64;
  v8f acc[8] = {};
  wave_gemm_32x64(A, lda, m0, W, ldb, n0, K, acc);
  const int rb = (lane >> 4) << 3, ci = lane & 15;
#pragma unroll
  for (int ti = 0; ti < 2; ti++)
#pragma unroll
    for (int nj = 0; nj < 4; nj++) {
      v8f c = acc[ti * 4 + nj];
#pragma unroll
      for (int r = 0; r < 8; r++) {
        int m = m0 + ti * 16 + rb + r;
        int n = n0 + nj * 16 + ci;
        float val = c[r] + (BIAS ? bias[n] : 0.f);
        if (ACT) val = val * sigm(val);
        long idx = (long)m * N + n;
        if (EPI == 0)      ((float*)outp)[idx] = val;
        else if (EPI == 1) ((u16*)outp)[idx] = f2bf(val);
        else               res[idx] += alpha * val;
      }
    }
}

// ---------------- attention kernels ----------------

// split qkv, add pos biases, rearrange to (b,h,t,hd) bf16 (and v transposed to (b,h,hd,t))
__global__ void qkv_prep_kernel(const float* qkv, const float* pbu, const float* pbv,
                                u16* qu, u16* qv, u16* kb, u16* vt) {
  long i = (long)blockIdx.x * 256 + threadIdx.x;   // over T*B*D
  if (i >= (long)conf::NR * conf::D) return;
  int dd = (int)(i & 511);
  long tb = i >> 9;
  int b = (int)(tb & 7);
  int t = (int)(tb >> 3);
  int h = dd >> 6, hd = dd & 63;
  const float* row = qkv + tb * 1536;
  float q = row[dd], k = row[512 + dd], v = row[1024 + dd];
  long bh = (long)b * 8 + h;
  qu[(bh * 512 + t) * 64 + hd] = f2bf(q + pbu[dd]);
  qv[(bh * 512 + t) * 64 + hd] = f2bf(q + pbv[dd]);
  kb[(bh * 512 + t) * 64 + hd] = f2bf(k);
  vt[(bh * 64 + hd) * 512 + t] = f2bf(v);
}

// bd term with fused rel_shift: scores[b,h,t,s] = (q+v)·p[T-1-t+s]  (initializes scores).
// p is padded to 1024 rows with row 1023 == 0; u=1023 results map to s>=512 and are
// rejected by the s-bounds check, so the whole kernel is branch-free in the GEMM loop.
__global__ void __launch_bounds__(256)
attn_bd_kernel(const u16* qvb, const u16* pb, float* scores) {
  const int z = blockIdx.z;                       // b*H + h
  const int lane = threadIdx.x & 31, w = threadIdx.x >> 5;
  const int m0 = blockIdx.y * 128 + (w & 3) * 32; // t
  const int n0 = blockIdx.x * 128 + (w >> 2) * 64;// u
  const u16* A  = qvb + (long)z * 512 * 64;       // lda 64
  const u16* Bp = pb + (z & 7) * 64;              // p row u for head h, ldb 512
  v8f acc[8] = {};
  wave_gemm_32x64(A, 64, m0, Bp, 512, n0, 64, acc);
  float* sc = scores + (long)z * 512 * 512;
  const int rb = (lane >> 4) << 3, ci = lane & 15;
#pragma unroll
  for (int ti = 0; ti < 2; ti++)
#pragma unroll
    for (int nj = 0; nj < 4; nj++) {
      v8f c = acc[ti * 4 + nj];
#pragma unroll
      for (int r = 0; r < 8; r++) {
        int t = m0 + ti * 16 + rb + r;
        int u = n0 + nj * 16 + ci;
        int s = u - 511 + t;                      // s = u - (T-1) + t
        if (s >= 0 && s < 512) sc[(long)t * 512 + s] = c[r];
      }
    }
}

// ac term: scores[b,h,t,s] += (q+u)·k[s]   (512x512, K=64, fully aligned)
__global__ void __launch_bounds__(256)
attn_ac_kernel(const u16* qub, const u16* kbb, float* scores) {
  const int z = blockIdx.z;
  const int lane = threadIdx.x & 31, w = threadIdx.x >> 5;
  const int m0 = blockIdx.y * 128 + (w & 3) * 32;
  const int n0 = blockIdx.x * 128 + (w >> 2) * 64;
  const u16* A  = qub + (long)z * 512 * 64;
  const u16* Bk = kbb + (long)z * 512 * 64;
  v8f acc[8] = {};
  wave_gemm_32x64(A, 64, m0, Bk, 64, n0, 64, acc);
  float* sc = scores + (long)z * 512 * 512;
  const int rb = (lane >> 4) << 3, ci = lane & 15;
#pragma unroll
  for (int ti = 0; ti < 2; ti++)
#pragma unroll
    for (int nj = 0; nj < 4; nj++) {
      v8f c = acc[ti * 4 + nj];
#pragma unroll
      for (int r = 0; r < 8; r++) {
        int t = m0 + ti * 16 + rb + r;
        int n = n0 + nj * 16 + ci;
        sc[(long)t * 512 + n] += c[r];
      }
    }
}

// wave-per-row softmax over 512 cols with 1/sqrt(hd) scale, emits bf16 attn
__global__ void softmax_kernel(const float* scores, u16* attn) {
  int wid = threadIdx.x >> 5, lane = threadIdx.x & 31;
  long row = (long)blockIdx.x * 8 + wid;          // rows = B*H*T = 32768
  const float* sr = scores + row * 512;
  float v[16], mx = -1e30f;
#pragma unroll
  for (int i = 0; i < 16; i++) { v[i] = sr[lane + 32 * i] * 0.125f; mx = fmaxf(mx, v[i]); }
#pragma unroll
  for (int m = 16; m >= 1; m >>= 1) mx = fmaxf(mx, __shfl_xor(mx, m, 32));
  float s = 0.f;
#pragma unroll
  for (int i = 0; i < 16; i++) { v[i] = __expf(v[i] - mx); s += v[i]; }
#pragma unroll
  for (int m = 16; m >= 1; m >>= 1) s += __shfl_xor(s, m, 32);
  float inv = 1.f / s;
#pragma unroll
  for (int i = 0; i < 16; i++) attn[row * 512 + lane + 32 * i] = f2bf(v[i] * inv);
}

// ctx = attn @ v, written bf16 directly in (t, b, h*64+hd) layout.
// N = 64 exactly -> 8 waves stacked down M (block tile 256x64), branch-free.
__global__ void __launch_bounds__(256)
attn_ctx_kernel(const u16* attn, const u16* vt, u16* ctx) {
  const int z = blockIdx.z;
  const int lane = threadIdx.x & 31, w = threadIdx.x >> 5;
  const int m0 = blockIdx.y * 256 + w * 32;       // t
  const u16* A  = attn + (long)z * 512 * 512;     // lda 512
  const u16* Bv = vt + (long)z * 64 * 512;        // row hd, ldb 512
  v8f acc[8] = {};
  wave_gemm_32x64(A, 512, m0, Bv, 512, 0, 512, acc);
  const int b = z >> 3, h = z & 7;
  const int rb = (lane >> 4) << 3, ci = lane & 15;
#pragma unroll
  for (int ti = 0; ti < 2; ti++)
#pragma unroll
    for (int nj = 0; nj < 4; nj++) {
      v8f c = acc[ti * 4 + nj];
#pragma unroll
      for (int r = 0; r < 8; r++) {
        int t = m0 + ti * 16 + rb + r;
        int hd = nj * 16 + ci;
        ctx[((long)t * 8 + b) * 512 + h * 64 + hd] = f2bf(c[r]);
      }
    }
}

// ---------------- conv-module elementwise kernels ----------------

__global__ void glu_kernel(const float* pw1o, float* out) {
  long i = (long)blockIdx.x * 256 + threadIdx.x;  // over T*B*D
  if (i >= (long)conf::NR * conf::D) return;
  int c = (int)(i & 511);
  long tb = i >> 9;
  float a = pw1o[tb * 1024 + c];
  float g = pw1o[tb * 1024 + 512 + c];
  out[i] = a * sigm(g);
}

__global__ void dwconv_bn_swish_kernel(const float* glu, const float* dww, const float* dwb,
                                       const float* bng, const float* bnb,
                                       const float* bnm, const float* bnv, u16* out) {
  long i = (long)blockIdx.x * 256 + threadIdx.x;  // over T*B*D
  if (i >= (long)conf::NR * conf::D) return;
  int c = (int)(i & 511);
  long tb = i >> 9;
  int b = (int)(tb & 7);
  int t = (int)(tb >> 3);
  float acc = 0.f;
#pragma unroll
  for (int j = 0; j < conf::KC; j++) {
    int tt = t + j - 15;
    if (tt >= 0 && tt < conf::T)
      acc += glu[((long)tt * 8 + b) * 512 + c] * dww[c * conf::KC + j];
  }
  acc += dwb[c];
  acc = (acc - bnm[c]) * rsqrtf(bnv[c] + 1e-5f) * bng[c] + bnb[c];
  acc = acc * sigm(acc);
  out[i] = f2bf(acc);
}

// ---------------- host orchestration ----------------

extern "C" void kernel_launch(void* const* d_in, const int* in_sizes, int n_in,
                              void* d_out, int out_size, void* d_ws, size_t ws_size,
                              hipStream_t stream) {
  using namespace conf;
  (void)in_sizes; (void)n_in; (void)out_size; (void)ws_size;

  const float* src      = (const float*)d_in[0];
  const float* pos_emb  = (const float*)d_in[1];
  const float* ffm_w1   = (const float*)d_in[2];
  const float* ffm_b1   = (const float*)d_in[3];
  const float* ffm_w2   = (const float*)d_in[4];
  const float* ffm_b2   = (const float*)d_in[5];
  const float* ff_w1    = (const float*)d_in[6];
  const float* ff_b1    = (const float*)d_in[7];
  const float* ff_w2    = (const float*)d_in[8];
  const float* ff_b2    = (const float*)d_in[9];
  const float* in_proj_w= (const float*)d_in[10];
  const float* in_proj_b= (const float*)d_in[11];
  const float* out_w    = (const float*)d_in[12];
  const float* out_b    = (const float*)d_in[13];
  const float* pos_w    = (const float*)d_in[14];
  const float* pbu      = (const float*)d_in[15];
  const float* pbv      = (const float*)d_in[16];
  const float* pw1_w    = (const float*)d_in[17];
  const float* pw1_b    = (const float*)d_in[18];
  const float* dw_w     = (const float*)d_in[19];
  const float* dw_b     = (const float*)d_in[20];
  const float* bn_g     = (const float*)d_in[21];
  const float* bn_b     = (const float*)d_in[22];
  const float* bn_m     = (const float*)d_in[23];
  const float* bn_v     = (const float*)d_in[24];
  const float* pw2_w    = (const float*)d_in[25];
  const float* pw2_b    = (const float*)d_in[26];
  const float* ln_ffm_g = (const float*)d_in[27];
  const float* ln_ffm_b = (const float*)d_in[28];
  const float* ln_mha_g = (const float*)d_in[29];
  const float* ln_mha_b = (const float*)d_in[30];
  const float* ln_conv_g= (const float*)d_in[31];
  const float* ln_conv_b= (const float*)d_in[32];
  const float* ln_ff_g  = (const float*)d_in[33];
  const float* ln_ff_b  = (const float*)d_in[34];
  const float* ln_fin_g = (const float*)d_in[35];
  const float* ln_fin_b = (const float*)d_in[36];

  char* ws = (char*)d_ws;
  auto U16 = [&](size_t off) { return (u16*)(ws + off); };
  auto F32 = [&](size_t off) { return (float*)(ws + off); };

  auto cvt = [&](const float* s, size_t off, int n) {
    f32_to_bf16_kernel<<<(n + 255) / 256, 256, 0, stream>>>(s, U16(off), n);
  };

  // ---- weights + pos_emb -> bf16 ----
  cvt(ffm_w1,    WB_FFM1, DFF * D);
  cvt(ffm_w2,    WB_FFM2, D * DFF);
  cvt(ff_w1,     WB_FF1,  DFF * D);
  cvt(ff_w2,     WB_FF2,  D * DFF);
  cvt(in_proj_w, WB_INP,  3 * D * D);
  cvt(out_w,     WB_OUT,  D * D);
  cvt(pos_w,     WB_POS,  D * D);
  cvt(pw1_w,     WB_PW1,  2 * D * D);
  cvt(pw2_w,     WB_PW2,  D * D);
  cvt(pos_emb,   PE_BF,   TP * D);
  // zero pad row 1023 of pos_emb so the pos GEMM / bd GEMM run branch-free
  zero_u16_kernel<<<2, 256, 0, stream>>>(U16(PE_BF) + (size_t)TP * D, D);

  const int nTok = NR * D;                         // 2,097,152
  copy_f32_kernel<<<(nTok + 255) / 256, 256, 0, stream>>>(src, F32(RES), nTok);

  // ---- macaron FF: res += 0.5 * W2(swish(W1(LN(res)))) ----
  layernorm_kernel<1><<<NR / 8, 256, 0, stream>>>(F32(RES), ln_ffm_g, ln_ffm_b, U16(XN), nullptr, NR);
  gemm_kernel<1, 1, 1><<<dim3(DFF / 128, NR / 128), 256, 0, stream>>>(
      U16(XN), D, U16(WB_FFM1), D, ffm_b1, U16(HID), nullptr, DFF, D, 1.f);
  gemm_kernel<2, 0, 1><<<dim3(D / 128, NR / 128), 256, 0, stream>>>(
      U16(HID), DFF, U16(WB_FFM2), DFF, ffm_b2, nullptr, F32(RES), D, DFF, 0.5f);

  // ---- rel-pos MHA ----
  layernorm_kernel<1><<<NR / 8, 256, 0, stream>>>(F32(RES), ln_mha_g, ln_mha_b, U16(XN), nullptr, NR);
  gemm_kernel<0, 0, 1><<<dim3(3 * D / 128, NR / 128), 256, 0, stream>>>(
      U16(XN), D, U16(WB_INP), D, in_proj_b, F32(QKV), nullptr, 3 * D, D, 1.f);
  qkv_prep_kernel<<<(nTok + 255) / 256, 256, 0, stream>>>(
      F32(QKV), pbu, pbv, U16(QU), U16(QV), U16(KB), U16(VT));
  // p = pos_emb @ pos_w^T, M padded to 1024 (row 1023 computes to 0 = pad row for bd)
  gemm_kernel<1, 0, 0><<<dim3(D / 128, 1024 / 128), 256, 0, stream>>>(
      U16(PE_BF), D, U16(WB_POS), D, nullptr, U16(PBF), nullptr, D, D, 1.f);
  attn_bd_kernel<<<dim3(1024 / 128, 4, BH), 256, 0, stream>>>(U16(QV), U16(PBF), F32(SC));
  attn_ac_kernel<<<dim3(4, 4, BH), 256, 0, stream>>>(U16(QU), U16(KB), F32(SC));
  softmax_kernel<<<BH * T / 8, 256, 0, stream>>>(F32(SC), U16(AT));
  attn_ctx_kernel<<<dim3(1, 2, BH), 256, 0, stream>>>(U16(AT), U16(VT), U16(CTX));
  gemm_kernel<2, 0, 1><<<dim3(D / 128, NR / 128), 256, 0, stream>>>(
      U16(CTX), D, U16(WB_OUT), D, out_b, nullptr, F32(RES), D, D, 1.f);

  // ---- conv module ----
  layernorm_kernel<1><<<NR / 8, 256, 0, stream>>>(F32(RES), ln_conv_g, ln_conv_b, U16(XN), nullptr, NR);
  gemm_kernel<0, 0, 1><<<dim3(2 * D / 128, NR / 128), 256, 0, stream>>>(
      U16(XN), D, U16(WB_PW1), D, pw1_b, F32(PW1O), nullptr, 2 * D, D, 1.f);
  glu_kernel<<<(nTok + 255) / 256, 256, 0, stream>>>(F32(PW1O), F32(GLUO));
  dwconv_bn_swish_kernel<<<(nTok + 255) / 256, 256, 0, stream>>>(
      F32(GLUO), dw_w, dw_b, bn_g, bn_b, bn_m, bn_v, U16(XN));
  gemm_kernel<2, 0, 1><<<dim3(D / 128, NR / 128), 256, 0, stream>>>(
      U16(XN), D, U16(WB_PW2), D, pw2_b, nullptr, F32(RES), D, D, 1.f);

  // ---- second FF: res += 0.5 * W2(swish(W1(LN(res)))) ----
  layernorm_kernel<1><<<NR / 8, 256, 0, stream>>>(F32(RES), ln_ff_g, ln_ff_b, U16(XN), nullptr, NR);
  gemm_kernel<1, 1, 1><<<dim3(DFF / 128, NR / 128), 256, 0, stream>>>(
      U16(XN), D, U16(WB_FF1), D, ff_b1, U16(HID), nullptr, DFF, D, 1.f);
  gemm_kernel<2, 0, 1><<<dim3(D / 128, NR / 128), 256, 0, stream>>>(
      U16(HID), DFF, U16(WB_FF2), DFF, ff_b2, nullptr, F32(RES), D, DFF, 0.5f);

  // ---- final layernorm -> d_out (fp32) ----
  layernorm_kernel<0><<<NR / 8, 256, 0, stream>>>(F32(RES), ln_fin_g, ln_fin_b, nullptr, (float*)d_out, NR);
}
